// COGMEN_32933809226562
// MI455X (gfx1250) — matrix operand order repaired and checked by reading, not compiled, not run on hardware
//
#include <hip/hip_runtime.h>
#include <hip/hip_bf16.h>
#include <math.h>

// ---------------- CDNA5 WMMA types ----------------
typedef __attribute__((ext_vector_type(16))) _Float16 v16h;
typedef __attribute__((ext_vector_type(8)))  float    v8f;
typedef __attribute__((ext_vector_type(8)))  _Float16 h8;

#define GF_BIAS  1
#define GF_RELU  2
#define GF_ACCUM 4
#define GF_C32   8
#define GF_CH    16

// ---------------- model constants ----------------
#define SS   256
#define BB   32
#define BS   8192            // B*S
#define DM   1380
#define DMP  1408            // DM padded to 32
#define DQ   4140            // 3*DM
#define NQG  4160            // DQ padded to 32
#define DFF  2048
#define DE   256
#define DH   256
#define NREL 8

// ============================================================
// B-operand packer: logical B[K,N] (f16, zero-padded to Kp x Ngrid) stored in
// WMMA-fragment order: chunk(nt, kb) of 512 f16, element offset inside chunk =
//   ((k&16)>>4)*256 + (n&15)*16 + (k&15)
// dst[((nt*KB + kb)*32 + lane)*16 + j]
// mode 0: B[k][n] = src[k*stride + off + n]
// mode 1: B[k][n] = src[n*stride + off + k]          (weight W[N,K] transpose)
// mode 2: B[k][n] = rel[((n>>8)*256 + k)*256 + (n&255)]   (RGCN relation stack)
// ============================================================
__global__ void k_pack_b(const float* __restrict__ src, long sBatch,
                         int Kreal, int Nreal, int stride, int off, int mode,
                         _Float16* __restrict__ dst, long dBatch, int KB, int Ngrid)
{
    src += (long)blockIdx.z * sBatch;
    dst += (long)blockIdx.z * dBatch;
    long idx = (long)blockIdx.x * 256 + threadIdx.x;
    long total = (long)KB * 512 * (Ngrid >> 4);
    if (idx >= total) return;
    int j    = (int)(idx & 15);
    int lane = (int)((idx >> 4) & 31);
    long blk = idx >> 9;
    int kb = (int)(blk % KB);
    int nt = (int)(blk / KB);
    int k = kb * 32 + (lane >> 4) * 16 + j;
    int n = nt * 16 + (lane & 15);
    float v = 0.f;
    if (k < Kreal && n < Nreal) {
        if (mode == 0)      v = src[(long)k * stride + off + n];
        else if (mode == 1) v = src[(long)n * stride + off + k];
        else                v = src[((long)(n >> 8) * 256 + k) * 256 + (n & 255)];
    }
    dst[idx] = (_Float16)v;
}

// ============================================================
// Wave-level WMMA GEMM:  C[M,N] = alpha * A[M,K] * Bp + (bias)(+C)(ReLU)
// A: f16 row-major (lda; K padded to mult of 64, 16B-aligned rows).
// Bp: fragment-packed (see k_pack_b).
// Block tile 128x64, BK=64. 8 waves as 4(M) x 2(N); each wave owns a 2x2 grid
// of 16x16 WMMA tiles (4 accumulators) -> 8 wmma per 64-K step per wave,
// 2x wmma/LDS ratio vs 1-tile waves. All tile traffic is b128.
// ============================================================
__launch_bounds__(256)
__global__ void k_gemm_wmma(const _Float16* __restrict__ A, int lda, long sA,
                            const _Float16* __restrict__ Bp, long sB,
                            float* __restrict__ C, int ldc, long sC,
                            _Float16* __restrict__ Ch, int ldch, long sCh,
                            const float* __restrict__ bias,
                            int M, int Nreal, int K, int flags, float alpha)
{
    __shared__ __align__(32) _Float16 AsF[8192];   // 2 kb x 8 m-tiles x 32 lanes x 16
    __shared__ __align__(32) _Float16 BsF[4096];   // 2 kb x 4 n-tiles x 32 lanes x 16

    const int z = blockIdx.z;
    A  += (long)z * sA;
    Bp += (long)z * sB;
    if (C)  C  += (long)z * sC;
    if (Ch) Ch += (long)z * sCh;

    const int tid  = threadIdx.x;
    const int wave = tid >> 5;
    const int lane = tid & 31;
    const int half = lane >> 4;
    const int l16  = lane & 15;

    const int bm  = blockIdx.y * 128;
    const int bn  = blockIdx.x * 64;
    const int wm  = wave >> 1;       // 0..3 -> rows wm*32 .. wm*32+31
    const int wn  = wave & 1;        // 0..1 -> cols wn*32 .. wn*32+31
    const int nt0 = blockIdx.x * 4;
    const int KB  = K >> 5;

    v8f acc00 = {}, acc01 = {}, acc10 = {}, acc11 = {};

    for (int k0 = 0; k0 < K; k0 += 64) {
        // ---- A tile 128x64: 4 x b128 per thread, fragment-swizzled LDS store
#pragma unroll
        for (int g = 0; g < 4; ++g) {
            int gid = g * 256 + tid;
            int r   = gid >> 3;        // row in tile (0..127)
            int cg  = gid & 7;         // 8-col group (0..7)
            int kbl = cg >> 2;         // k-block 0/1
            int cgl = cg & 3;          // col group within k-block
            int gm  = bm + r;
            h8 v = {};
            if (gm < M)
                v = *(const h8*)(A + (long)gm * lda + k0 + kbl * 32 + cgl * 8);
            *(h8*)(AsF + kbl * 4096 +
                   (((r >> 4) * 32 + (cgl >> 1) * 16 + (r & 15)) << 4) + (cgl & 1) * 8) = v;
        }
        // ---- B tile: block copy of 8 packed chunks (2 x b128 per thread)
#pragma unroll
        for (int g = 0; g < 2; ++g) {
            int slot   = g * 256 + tid;
            int chunk  = slot >> 6;        // 0..7
            int within = slot & 63;
            int kbl = chunk >> 2, ntl = chunk & 3;
            const _Float16* sB1 = Bp + (((long)(nt0 + ntl) * KB + ((k0 >> 5) + kbl)) << 9)
                                  + within * 8;
            *(h8*)(BsF + kbl * 2048 + ntl * 512 + within * 8) = *(const h8*)sB1;
        }
        __syncthreads();

#pragma unroll
        for (int kk = 0; kk < 2; ++kk) {
            v16h af0 = *(const v16h*)(AsF + kk * 4096 + ((wm * 2 + 0) << 9) + lane * 16);
            v16h af1 = *(const v16h*)(AsF + kk * 4096 + ((wm * 2 + 1) << 9) + lane * 16);
            v16h bf0 = *(const v16h*)(BsF + kk * 2048 + ((wn * 2 + 0) << 9) + lane * 16);
            v16h bf1 = *(const v16h*)(BsF + kk * 2048 + ((wn * 2 + 1) << 9) + lane * 16);
            acc00 = __builtin_amdgcn_wmma_f32_16x16x32_f16(false, af0, false, bf0,
                                                           (short)0, acc00, false, false);
            acc01 = __builtin_amdgcn_wmma_f32_16x16x32_f16(false, af0, false, bf1,
                                                           (short)0, acc01, false, false);
            acc10 = __builtin_amdgcn_wmma_f32_16x16x32_f16(false, af1, false, bf0,
                                                           (short)0, acc10, false, false);
            acc11 = __builtin_amdgcn_wmma_f32_16x16x32_f16(false, af1, false, bf1,
                                                           (short)0, acc11, false, false);
        }
        __syncthreads();
    }

    // epilogue: C/D layout -> lane holds col n, VGPR r holds row r + 8*half
#pragma unroll
    for (int tm = 0; tm < 2; ++tm) {
#pragma unroll
        for (int tn = 0; tn < 2; ++tn) {
            v8f a = (tm == 0) ? (tn == 0 ? acc00 : acc01)
                              : (tn == 0 ? acc10 : acc11);
            int n = bn + wn * 32 + tn * 16 + l16;
            float bv = ((flags & GF_BIAS) && n < Nreal) ? bias[n] : 0.f;
#pragma unroll
            for (int r = 0; r < 8; ++r) {
                int m = bm + wm * 32 + tm * 16 + r + half * 8;
                float v = alpha * a[r] + bv;
                if ((flags & GF_ACCUM) && m < M && n < Nreal) v += C[(long)m * ldc + n];
                if (flags & GF_RELU) v = v > 0.f ? v : 0.f;
                if ((flags & GF_C32) && m < M && n < Nreal) C[(long)m * ldc + n] = v;
                if (flags & GF_CH) {
                    _Float16 hv = (m < M && n < Nreal) ? (_Float16)v : (_Float16)0.f;
                    if (n < ldch) Ch[(long)m * ldch + n] = hv;
                }
            }
        }
    }
}

// concat(textf,acouf,visuf) -> X f32 [BS,DM] + Xh f16 [BS,DMP]
__global__ void k_concat(const float* __restrict__ tf, const float* __restrict__ af,
                         const float* __restrict__ vf,
                         float* __restrict__ X, _Float16* __restrict__ Xh)
{
    long idx = (long)blockIdx.x * 256 + threadIdx.x;
    if (idx >= (long)BS * DMP) return;
    int r = (int)(idx / DMP), d = (int)(idx % DMP);
    int b = r >> 8, s = r & 255;
    float v = 0.f;
    if (d < 768)       v = tf[((long)s * BB + b) * 768 + d];
    else if (d < 868)  v = af[((long)s * BB + b) * 100 + (d - 768)];
    else if (d < DM)   v = vf[((long)s * BB + b) * 512 + (d - 868)];
    Xh[idx] = (_Float16)v;
    if (d < DM) X[(long)r * DM + d] = v;
}

// f32 [rows, cols] (row stride, col offset) -> f16 [rows, ldd] zero-padded (A operands)
__global__ void k_cvt_pad(const float* __restrict__ src, long sBatch, int rows, int cols,
                          int stride, int off, _Float16* __restrict__ dst, long dBatch, int ldd)
{
    src += (long)blockIdx.z * sBatch;
    dst += (long)blockIdx.z * dBatch;
    long idx = (long)blockIdx.x * 256 + threadIdx.x;
    if (idx >= (long)rows * ldd) return;
    int r = (int)(idx / ldd), c = (int)(idx % ldd);
    dst[idx] = (c < cols) ? (_Float16)src[(long)r * stride + off + c] : (_Float16)0.f;
}

// out = LN(x + y) over D=1380; also f16 padded copy
__global__ void k_add_ln(const float* __restrict__ x, const float* __restrict__ y,
                         const float* __restrict__ g, const float* __restrict__ b,
                         float* __restrict__ out, _Float16* __restrict__ outh)
{
    const int r = blockIdx.x, tid = threadIdx.x;
    __shared__ float rs[256], rq[256];
    float s = 0.f, q = 0.f;
    for (int d = tid; d < DM; d += 256) {
        float v = x[(long)r * DM + d] + y[(long)r * DM + d];
        s += v; q += v * v;
    }
    rs[tid] = s; rq[tid] = q;
    __syncthreads();
    for (int o = 128; o > 0; o >>= 1) {
        if (tid < o) { rs[tid] += rs[tid + o]; rq[tid] += rq[tid + o]; }
        __syncthreads();
    }
    float mu  = rs[0] / (float)DM;
    float var = rq[0] / (float)DM - mu * mu;
    float inv = rsqrtf(var + 1e-5f);
    for (int d = tid; d < DMP; d += 256) {
        if (d < DM) {
            float v = x[(long)r * DM + d] + y[(long)r * DM + d];
            v = g[d] * (v - mu) * inv + b[d];
            out[(long)r * DM + d] = v;
            outh[(long)r * DMP + d] = (_Float16)v;
        } else {
            outh[(long)r * DMP + d] = (_Float16)0.f;
        }
    }
}

// row softmax over 256, emit f16 probs
__global__ void k_softmax(const float* __restrict__ sc, _Float16* __restrict__ ph)
{
    const int r = blockIdx.x, tid = threadIdx.x;
    __shared__ float red[256];
    float v = sc[(long)r * 256 + tid];
    red[tid] = v; __syncthreads();
    for (int o = 128; o > 0; o >>= 1) {
        if (tid < o) red[tid] = fmaxf(red[tid], red[tid + o]);
        __syncthreads();
    }
    float m = red[0]; __syncthreads();
    float e = expf(v - m);
    red[tid] = e; __syncthreads();
    for (int o = 128; o > 0; o >>= 1) {
        if (tid < o) red[tid] += red[tid + o];
        __syncthreads();
    }
    ph[(long)r * 256 + tid] = (_Float16)(e / red[0]);
}

__global__ void k_gather(const float* __restrict__ xp, const int* __restrict__ nidx,
                         int N, int Np, float* __restrict__ nodes, _Float16* __restrict__ nodesh)
{
    long idx = (long)blockIdx.x * 256 + threadIdx.x;
    if (idx >= (long)Np * DE) return;
    int r = (int)(idx >> 8), c = (int)(idx & 255);
    float v = (r < N) ? xp[(long)nidx[r] * DE + c] : 0.f;
    nodes[idx] = v;
    nodesh[idx] = (_Float16)v;
}

__global__ void k_edge_count(const int* __restrict__ dst, const int* __restrict__ et,
                             float* __restrict__ cnt, int E)
{
    int e = blockIdx.x * 256 + threadIdx.x;
    if (e < E) atomicAdd(&cnt[(long)dst[e] * NREL + et[e]], 1.f);
}

__global__ void k_rgcn_scatter(const float* __restrict__ xr, const int* __restrict__ src,
                               const int* __restrict__ dst, const int* __restrict__ et,
                               const float* __restrict__ cnt, float* __restrict__ hacc)
{
    int e = blockIdx.x, c = threadIdx.x;
    int s = src[e], d = dst[e], t = et[e];
    float w = 1.f / fmaxf(cnt[(long)d * NREL + t], 1.f);
    atomicAdd(&hacc[(long)d * DH + c], xr[(long)s * (NREL * DH) + t * DH + c] * w);
}

__global__ void k_edge_logit(const float* __restrict__ qn, const float* __restrict__ kn,
                             const int* __restrict__ src, const int* __restrict__ dst,
                             float* __restrict__ logit, int E)
{
    int e = blockIdx.x * 8 + (threadIdx.x >> 5);
    int lane = threadIdx.x & 31;
    if (e >= E) return;
    int s = src[e], d = dst[e];
    float acc = 0.f;
    for (int i = lane; i < DH; i += 32)
        acc += qn[(long)d * DH + i] * kn[(long)s * DH + i];
    for (int o = 16; o > 0; o >>= 1) acc += __shfl_xor(acc, o, 32);
    if (lane == 0) logit[e] = acc * (1.f / 16.f);   // 1/sqrt(256)
}

__device__ __forceinline__ unsigned f_key(float f)
{
    unsigned u = __float_as_uint(f);
    return (u >> 31) ? ~u : (u | 0x80000000u);
}
__device__ __forceinline__ float f_dec(unsigned k)
{
    return (k & 0x80000000u) ? __uint_as_float(k ^ 0x80000000u) : __uint_as_float(~k);
}

__global__ void k_fill_u32(unsigned* __restrict__ p, unsigned v, long n)
{
    long i = (long)blockIdx.x * 256 + threadIdx.x;
    if (i < n) p[i] = v;
}

__global__ void k_seg_max(const float* __restrict__ logit, const int* __restrict__ dst,
                          unsigned* __restrict__ emax, int E)
{
    int e = blockIdx.x * 256 + threadIdx.x;
    if (e < E) atomicMax(&emax[dst[e]], f_key(logit[e]));
}

__global__ void k_edge_exp(const float* __restrict__ logit, const int* __restrict__ dst,
                           const unsigned* __restrict__ emax, float* __restrict__ eexp,
                           float* __restrict__ esum, int E)
{
    int e = blockIdx.x * 256 + threadIdx.x;
    if (e >= E) return;
    float m = f_dec(emax[dst[e]]);
    float v = expf(logit[e] - m);
    eexp[e] = v;
    atomicAdd(&esum[dst[e]], v);
}

__global__ void k_attn_scatter(const float* __restrict__ eexp, const float* __restrict__ esum,
                               const int* __restrict__ src, const int* __restrict__ dst,
                               const float* __restrict__ vn, float* __restrict__ hout)
{
    int e = blockIdx.x, c = threadIdx.x;
    float a = eexp[e] / esum[dst[e]];
    atomicAdd(&hout[(long)dst[e] * DH + c], a * vn[(long)src[e] * DH + c]);
}

__global__ void k_bn_stats(const float* __restrict__ h, int N,
                           float* __restrict__ mu, float* __restrict__ var)
{
    const int c = blockIdx.x, tid = threadIdx.x;
    __shared__ float rs[256], rq[256];
    float s = 0.f, q = 0.f;
    for (int r = tid; r < N; r += 256) {
        float v = h[(long)r * DH + c];
        s += v; q += v * v;
    }
    rs[tid] = s; rq[tid] = q;
    __syncthreads();
    for (int o = 128; o > 0; o >>= 1) {
        if (tid < o) { rs[tid] += rs[tid + o]; rq[tid] += rq[tid + o]; }
        __syncthreads();
    }
    if (tid == 0) {
        float m = rs[0] / (float)N;
        mu[c] = m;
        var[c] = rq[0] / (float)N - m * m;
    }
}

__global__ void k_bn_apply(const float* __restrict__ h, const float* __restrict__ mu,
                           const float* __restrict__ var, const float* __restrict__ g,
                           const float* __restrict__ b, int N, int Np,
                           _Float16* __restrict__ h2h)
{
    long idx = (long)blockIdx.x * 256 + threadIdx.x;
    if (idx >= (long)Np * DH) return;
    int r = (int)(idx >> 8), c = (int)(idx & 255);
    float o = 0.f;
    if (r < N) {
        float v = h[idx];
        v = g[c] * (v - mu[c]) * rsqrtf(var[c] + 1e-5f) + b[c];
        o = v > 0.f ? v : 0.01f * v;
    }
    h2h[idx] = (_Float16)o;
}

__global__ void k_logsoftmax(const float* __restrict__ z, int N, float* __restrict__ out)
{
    int i = blockIdx.x * 256 + threadIdx.x;
    if (i >= N) return;
    float v[6], m = -1e30f;
#pragma unroll
    for (int j = 0; j < 6; ++j) { v[j] = z[(long)i * 16 + j]; m = fmaxf(m, v[j]); }
    float s = 0.f;
#pragma unroll
    for (int j = 0; j < 6; ++j) s += expf(v[j] - m);
    float l = m + logf(s);
#pragma unroll
    for (int j = 0; j < 6; ++j) out[(long)i * 6 + j] = v[j] - l;
}

// ============================================================
extern "C" void kernel_launch(void* const* d_in, const int* in_sizes, int n_in,
                              void* d_out, int out_size, void* d_ws, size_t ws_size,
                              hipStream_t stream)
{
    const float* textf = (const float*)d_in[0];
    const float* acouf = (const float*)d_in[1];
    const float* visuf = (const float*)d_in[2];
    const int*   nidx  = (const int*)d_in[3];
    const int*   eidx  = (const int*)d_in[4];
    const int*   etyp  = (const int*)d_in[5];
    const float* in_proj_w = (const float*)d_in[6];
    const float* in_proj_b = (const float*)d_in[7];
    const float* out_proj_w = (const float*)d_in[8];
    const float* out_proj_b = (const float*)d_in[9];
    const float* ln1_g = (const float*)d_in[10];
    const float* ln1_b = (const float*)d_in[11];
    const float* ff1_w = (const float*)d_in[12];
    const float* ff1_b = (const float*)d_in[13];
    const float* ff2_w = (const float*)d_in[14];
    const float* ff2_b = (const float*)d_in[15];
    const float* ln2_g = (const float*)d_in[16];
    const float* ln2_b = (const float*)d_in[17];
    const float* proj_w = (const float*)d_in[18];
    const float* proj_b = (const float*)d_in[19];
    const float* rgcn_rel  = (const float*)d_in[20];
    const float* rgcn_root = (const float*)d_in[21];
    const float* rgcn_b    = (const float*)d_in[22];
    const float* tc_q_w = (const float*)d_in[23];
    const float* tc_q_b = (const float*)d_in[24];
    const float* tc_k_w = (const float*)d_in[25];
    const float* tc_k_b = (const float*)d_in[26];
    const float* tc_v_w = (const float*)d_in[27];
    const float* tc_v_b = (const float*)d_in[28];
    const float* tc_skip_w = (const float*)d_in[29];
    const float* tc_skip_b = (const float*)d_in[30];
    const float* bn_g = (const float*)d_in[31];
    const float* bn_b = (const float*)d_in[32];
    const float* cls1_w = (const float*)d_in[33];
    const float* cls1_b = (const float*)d_in[34];
    const float* cls2_w = (const float*)d_in[35];
    const float* cls2_b = (const float*)d_in[36];
    float* outp = (float*)d_out;

    const int N  = in_sizes[3];
    const int E  = in_sizes[4] / 2;
    const int Np = ((N + 127) / 128) * 128;   // GEMM M tile = 128
    const int* srcp = eidx;
    const int* dstp = eidx + E;

    // ---- workspace carve-out ----
    size_t off = 0;
    auto take = [&](size_t bytes) -> void* {
        void* p = (char*)d_ws + off;
        off += (bytes + 255) & ~(size_t)255;
        return p;
    };
    float*     X     = (float*)take((size_t)BS * DM * 4);
    _Float16*  XH    = (_Float16*)take((size_t)BS * DMP * 2);
    float*     QKV   = (float*)take((size_t)BS * DQ * 4);
    _Float16*  BW    = (_Float16*)take((size_t)DMP * NQG * 2);   // packed-B staging (max)
    _Float16*  QH    = (_Float16*)take((size_t)BS * DMP * 2);
    _Float16*  KTH   = (_Float16*)take((size_t)BB * DMP * SS * 2);   // packed per batch
    _Float16*  VH    = (_Float16*)take((size_t)BB * SS * DMP * 2);   // packed per batch
    float*     SC    = (float*)take((size_t)BB * SS * SS * 4);
    _Float16*  PRH   = (_Float16*)take((size_t)BB * SS * SS * 2);
    _Float16*  ATH   = (_Float16*)take((size_t)BS * DMP * 2);
    float*     TMP1  = (float*)take((size_t)BS * DM * 4);
    float*     X1    = (float*)take((size_t)BS * DM * 4);
    _Float16*  X1H   = (_Float16*)take((size_t)BS * DMP * 2);
    _Float16*  FFH   = (_Float16*)take((size_t)BS * DFF * 2);
    float*     X2    = (float*)take((size_t)BS * DM * 4);
    _Float16*  X2H   = (_Float16*)take((size_t)BS * DMP * 2);
    float*     XP    = (float*)take((size_t)BS * DE * 4);
    float*     NODES = (float*)take((size_t)BS * DE * 4);
    _Float16*  NODESH= (_Float16*)take((size_t)BS * DE * 2);
    float*     XR    = (float*)take((size_t)BS * NREL * DH * 4);
    float*     CNT   = (float*)take((size_t)BS * NREL * 4);
    float*     HACC  = (float*)take((size_t)BS * DH * 4);
    _Float16*  HH    = (_Float16*)take((size_t)BS * DH * 2);
    float*     QN    = (float*)take((size_t)BS * DH * 4);
    float*     KN    = (float*)take((size_t)BS * DH * 4);
    float*     VN    = (float*)take((size_t)BS * DH * 4);
    float*     LOGIT = (float*)take((size_t)262144 * 4);
    float*     EEXP  = (float*)take((size_t)262144 * 4);
    unsigned*  EMAXU = (unsigned*)take((size_t)BS * 4);
    float*     ESUM  = (float*)take((size_t)BS * 4);
    float*     HOUT  = (float*)take((size_t)BS * DH * 4);
    float*     MU    = (float*)take(256 * 4);
    float*     VARV  = (float*)take(256 * 4);
    _Float16*  H2H   = (_Float16*)take((size_t)BS * DH * 2);
    _Float16*  Z1H   = (_Float16*)take((size_t)BS * DH * 2);
    float*     ZL    = (float*)take((size_t)BS * 16 * 4);
    (void)ws_size; (void)n_in; (void)out_size;

    auto cdiv = [](long a, long b) { return (int)((a + b - 1) / b); };

    // pack B operand (batched): logical [Kp x Ngrid] from f32 source
    auto packB = [&](const float* src, long sBatch, int Kreal, int Nreal, int stride,
                     int soff, int mode, _Float16* dst, long dBatch, int Kp, int Ngrid,
                     int batch) {
        dim3 g(cdiv((long)Kp * Ngrid, 256), 1, batch);
        k_pack_b<<<g, 256, 0, stream>>>(src, sBatch, Kreal, Nreal, stride, soff, mode,
                                        dst, dBatch, Kp >> 5, Ngrid);
    };

    auto gemm = [&](const _Float16* A, int lda, long sA,
                    const _Float16* Bp, long sB,
                    float* C, int ldc, long sC,
                    _Float16* Ch, int ldch, long sCh,
                    const float* bias, int M, int Mp, int Nreal, int Ngrid, int K,
                    int flags, float alpha, int batch) {
        dim3 g(Ngrid / 64, Mp / 128, batch);
        k_gemm_wmma<<<g, 256, 0, stream>>>(A, lda, sA, Bp, sB,
                                           C, ldc, sC, Ch, ldch, sCh,
                                           bias, M, Nreal, K, flags, alpha);
    };

    // ===== SeqContext transformer =====
    k_concat<<<cdiv((long)BS * DMP, 256), 256, 0, stream>>>(textf, acouf, visuf, X, XH);

    // qkv = X @ in_proj_w^T + b
    packB(in_proj_w, 0, DM, DQ, DM, 0, 1, BW, 0, DMP, NQG, 1);
    gemm(XH, DMP, 0, BW, 0, QKV, DQ, 0, nullptr, 0, 0, in_proj_b,
         BS, BS, DQ, NQG, DMP, GF_BIAS | GF_C32, 1.f, 1);

    // q (f16 padded A), k^T packed-B per batch, v packed-B per batch
    k_cvt_pad<<<cdiv((long)BS * DMP, 256), 256, 0, stream>>>(QKV, 0, BS, DM, DQ, 0, QH, 0, DMP);
    packB(QKV, (long)SS * DQ, DM, SS, DQ, DM, 1, KTH, (long)DMP * SS, DMP, SS, BB);
    packB(QKV, (long)SS * DQ, SS, DM, DQ, 2 * DM, 0, VH, (long)SS * DMP, SS, DMP, BB);

    // scores = q k^T / sqrt(DM)   (batched)
    gemm(QH, DMP, (long)SS * DMP, KTH, (long)DMP * SS,
         SC, SS, (long)SS * SS, nullptr, 0, 0, nullptr,
         SS, SS, SS, SS, DMP, GF_C32, rsqrtf((float)DM), BB);

    k_softmax<<<BS, 256, 0, stream>>>(SC, PRH);

    // attn = probs @ v  (f16 out only, batched)
    gemm(PRH, SS, (long)SS * SS, VH, (long)SS * DMP,
         nullptr, 0, 0, ATH, DMP, (long)SS * DMP, nullptr,
         SS, SS, DM, DMP, SS, GF_CH, 1.f, BB);

    // out_proj
    packB(out_proj_w, 0, DM, DM, DM, 0, 1, BW, 0, DMP, DMP, 1);
    gemm(ATH, DMP, 0, BW, 0, TMP1, DM, 0, nullptr, 0, 0, out_proj_b,
         BS, BS, DM, DMP, DMP, GF_BIAS | GF_C32, 1.f, 1);

    k_add_ln<<<BS, 256, 0, stream>>>(X, TMP1, ln1_g, ln1_b, X1, X1H);

    // FFN
    packB(ff1_w, 0, DM, DFF, DM, 0, 1, BW, 0, DMP, DFF, 1);
    gemm(X1H, DMP, 0, BW, 0, nullptr, 0, 0, FFH, DFF, 0, ff1_b,
         BS, BS, DFF, DFF, DMP, GF_BIAS | GF_RELU | GF_CH, 1.f, 1);
    packB(ff2_w, 0, DFF, DM, DFF, 0, 1, BW, 0, DFF, DMP, 1);
    gemm(FFH, DFF, 0, BW, 0, TMP1, DM, 0, nullptr, 0, 0, ff2_b,
         BS, BS, DM, DMP, DFF, GF_BIAS | GF_C32, 1.f, 1);

    k_add_ln<<<BS, 256, 0, stream>>>(X1, TMP1, ln2_g, ln2_b, X2, X2H);

    // proj to D_E
    packB(proj_w, 0, DM, DE, DM, 0, 1, BW, 0, DMP, DE, 1);
    gemm(X2H, DMP, 0, BW, 0, XP, DE, 0, nullptr, 0, 0, proj_b,
         BS, BS, DE, DE, DMP, GF_BIAS | GF_C32, 1.f, 1);

    // ===== graph phase =====
    k_gather<<<cdiv((long)Np * DE, 256), 256, 0, stream>>>(XP, nidx, N, Np, NODES, NODESH);

    // RGCN: xr[n, r*256+h] = nodes @ rel (all relations in one GEMM)
    packB(rgcn_rel, 0, DE, NREL * DH, 0, 0, 2, BW, 0, DE, NREL * DH, 1);
    gemm(NODESH, DE, 0, BW, 0, XR, NREL * DH, 0, nullptr, 0, 0, nullptr,
         N, Np, NREL * DH, NREL * DH, DE, GF_C32, 1.f, 1);

    hipMemsetAsync(CNT, 0, (size_t)BS * NREL * 4, stream);
    k_edge_count<<<cdiv(E, 256), 256, 0, stream>>>(dstp, etyp, CNT, E);
    hipMemsetAsync(HACC, 0, (size_t)BS * DH * 4, stream);
    k_rgcn_scatter<<<E, 256, 0, stream>>>(XR, srcp, dstp, etyp, CNT, HACC);

    // h = segsum + nodes@root + bias  (ACCUM), also f16 copy
    packB(rgcn_root, 0, DE, DH, DH, 0, 0, BW, 0, DE, DH, 1);
    gemm(NODESH, DE, 0, BW, 0, HACC, DH, 0, HH, DH, 0, rgcn_b,
         N, Np, DH, DH, DE, GF_BIAS | GF_ACCUM | GF_C32 | GF_CH, 1.f, 1);

    // TransformerConv q/k/v
    packB(tc_q_w, 0, DH, DH, DH, 0, 1, BW, 0, DH, DH, 1);
    gemm(HH, DH, 0, BW, 0, QN, DH, 0, nullptr, 0, 0, tc_q_b,
         N, Np, DH, DH, DH, GF_BIAS | GF_C32, 1.f, 1);
    packB(tc_k_w, 0, DH, DH, DH, 0, 1, BW, 0, DH, DH, 1);
    gemm(HH, DH, 0, BW, 0, KN, DH, 0, nullptr, 0, 0, tc_k_b,
         N, Np, DH, DH, DH, GF_BIAS | GF_C32, 1.f, 1);
    packB(tc_v_w, 0, DH, DH, DH, 0, 1, BW, 0, DH, DH, 1);
    gemm(HH, DH, 0, BW, 0, VN, DH, 0, nullptr, 0, 0, tc_v_b,
         N, Np, DH, DH, DH, GF_BIAS | GF_C32, 1.f, 1);

    // edge softmax attention
    k_edge_logit<<<cdiv(E, 8), 256, 0, stream>>>(QN, KN, srcp, dstp, LOGIT, E);
    k_fill_u32<<<cdiv(BS, 256), 256, 0, stream>>>(EMAXU, 0u, BS);
    k_seg_max<<<cdiv(E, 256), 256, 0, stream>>>(LOGIT, dstp, EMAXU, E);
    hipMemsetAsync(ESUM, 0, (size_t)BS * 4, stream);
    k_edge_exp<<<cdiv(E, 256), 256, 0, stream>>>(LOGIT, dstp, EMAXU, EEXP, ESUM, E);
    hipMemsetAsync(HOUT, 0, (size_t)BS * DH * 4, stream);
    k_attn_scatter<<<E, 256, 0, stream>>>(EEXP, ESUM, srcp, dstp, VN, HOUT);

    // + h @ skip^T + bias (ACCUM)
    packB(tc_skip_w, 0, DH, DH, DH, 0, 1, BW, 0, DH, DH, 1);
    gemm(HH, DH, 0, BW, 0, HOUT, DH, 0, nullptr, 0, 0, tc_skip_b,
         N, Np, DH, DH, DH, GF_BIAS | GF_ACCUM | GF_C32, 1.f, 1);

    // BatchNorm + leaky relu
    k_bn_stats<<<DH, 256, 0, stream>>>(HOUT, N, MU, VARV);
    k_bn_apply<<<cdiv((long)Np * DH, 256), 256, 0, stream>>>(HOUT, MU, VARV, bn_g, bn_b, N, Np, H2H);

    // classifier
    packB(cls1_w, 0, DH, DH, DH, 0, 1, BW, 0, DH, DH, 1);
    gemm(H2H, DH, 0, BW, 0, nullptr, 0, 0, Z1H, DH, 0, cls1_b,
         N, Np, DH, DH, DH, GF_BIAS | GF_RELU | GF_CH, 1.f, 1);
    packB(cls2_w, 0, DH, 6, DH, 0, 1, BW, 0, DH, 64, 1);
    gemm(Z1H, DH, 0, BW, 0, ZL, 16, 0, nullptr, 0, 0, cls2_b,
         N, Np, 6, 64, DH, GF_BIAS | GF_C32, 1.f, 1);

    k_logsoftmax<<<cdiv(N, 256), 256, 0, stream>>>(ZL, N, outp);
}
